// AttentionDecoder_54846732370359
// MI455X (gfx1250) — matrix operand order repaired and checked by reading, not compiled, not run on hardware
//
#include <hip/hip_runtime.h>
#include <stdint.h>

// Problem dims (fixed by the reference)
#define B_   128
#define S_   256
#define T_   24
#define H_   1024
#define A_   1024
#define G4H  4096          // 4*H
#define KCAT 3072          // H + A + H  (xin = [x | context | h])

typedef __attribute__((ext_vector_type(16))) __bf16 v16bf;
typedef __attribute__((ext_vector_type(8)))  float  v8f;

union FragBF { uint32_t u[8]; v16bf v; };
union PackBF { __bf16 h[2]; uint32_t u; };

__device__ __forceinline__ float sigmoidf_(float x) {
    return 1.0f / (1.0f + __expf(-x));
}
__device__ __forceinline__ float fast_tanh(float x) {
    // tanh(x) = 1 - 2/(exp(2x)+1); saturates correctly at +-1 for |x| large
    float e = __expf(2.0f * x);
    return 1.0f - 2.0f / (e + 1.0f);
}

// ---------------------------------------------------------------------------
// C[M,N] (f32) = A[M,K](bf16 row-major) @ B[N,K](bf16 row-major)^T + bias[N]
// Block: 256 threads = 8 waves arranged 2(M) x 4(N).
//   MT=1: wave tile 16x64, block tile 32x256  (small-M, occupancy-bound GEMMs)
//   MT=2: wave tile 32x64, block tile 64x256  (tall GEMM: 2x B-fragment reuse)
// Requires: M % (32*MT)==0, N%256==0, K%32==0 (true for all call sites here).
// A/B fragments (16x32, 32x16) use contiguous k = 16*(lane/16) + s, not the ISA 7.12.2 table order; f32 C/D per ISA.
// ---------------------------------------------------------------------------
template<int MT>
__global__ __launch_bounds__(256)
void gemm_bf16_nt(const uint32_t* __restrict__ A,   // bf16 viewed as dwords
                  const uint32_t* __restrict__ Bm,  // bf16 viewed as dwords
                  const float*    __restrict__ bias,
                  float*          __restrict__ C,
                  int M, int N, int K)
{
    const int lane = threadIdx.x & 31;
    const int wave = threadIdx.x >> 5;
    const int lm   = lane & 15;        // 0..15
    const int g    = lane >> 4;        // lane group 0/1
    const int m0   = blockIdx.y * (32 * MT) + (wave >> 2) * (16 * MT);
    const int n0   = blockIdx.x * 256 + (wave & 3) * 64;
    const int Kd   = K >> 1;           // dwords per row

    v8f acc[MT][4];
    #pragma unroll
    for (int mt = 0; mt < MT; ++mt)
        #pragma unroll
        for (int nt = 0; nt < 4; ++nt)
            acc[mt][nt] = (v8f){};

    const uint32_t* arow[MT];
    #pragma unroll
    for (int mt = 0; mt < MT; ++mt)
        arow[mt] = A + (size_t)(m0 + 16 * mt + lm) * Kd;
    const uint32_t* brow[4];
    #pragma unroll
    for (int nt = 0; nt < 4; ++nt)
        brow[nt] = Bm + (size_t)(n0 + 16 * nt + lm) * Kd + 8 * g;

    for (int k0 = 0; k0 < K; k0 += 32) {
        const int kd = k0 >> 1;

        // A fragment: lanes 0-15 hold K=0..15 (v0-7);
        //             lanes 16-31 hold K=16..31.
        FragBF fa[MT];
        #pragma unroll
        for (int mt = 0; mt < MT; ++mt) {
            #pragma unroll
            for (int j = 0; j < 4; ++j) fa[mt].u[j]     = arow[mt][kd + 8 * g + j];
            #pragma unroll
            for (int j = 0; j < 4; ++j) fa[mt].u[4 + j] = arow[mt][kd + 4 + 8 * g + j];
        }

        // B fragments: lane holds column n; lanes 0-15 K=0..15, 16-31 K=16..31
        #pragma unroll
        for (int nt = 0; nt < 4; ++nt) {
            FragBF fb;
            #pragma unroll
            for (int j = 0; j < 8; ++j) fb.u[j] = brow[nt][kd + j];
            // Prefetch the next K-tile of this B row (lowers to global_prefetch_b8)
            __builtin_prefetch(brow[nt] + kd + 16, 0, 1);
            #pragma unroll
            for (int mt = 0; mt < MT; ++mt)
                acc[mt][nt] = __builtin_amdgcn_wmma_f32_16x16x32_bf16(
                    false, fa[mt].v, false, fb.v, (short)0, acc[mt][nt],
                    false, false);
        }
    }

    // C/D layout: VGPR r -> M = m0 + 16*mt + r + 8*g, N = n0 + nt*16 + lm
    #pragma unroll
    for (int mt = 0; mt < MT; ++mt) {
        #pragma unroll
        for (int r = 0; r < 8; ++r) {
            const int m = m0 + 16 * mt + r + 8 * g;
            float* crow = C + (size_t)m * N;
            if (bias) {
                crow[n0 +  0 + lm] = acc[mt][0][r] + bias[n0 +  0 + lm];
                crow[n0 + 16 + lm] = acc[mt][1][r] + bias[n0 + 16 + lm];
                crow[n0 + 32 + lm] = acc[mt][2][r] + bias[n0 + 32 + lm];
                crow[n0 + 48 + lm] = acc[mt][3][r] + bias[n0 + 48 + lm];
            } else {
                crow[n0 +  0 + lm] = acc[mt][0][r];
                crow[n0 + 16 + lm] = acc[mt][1][r];
                crow[n0 + 32 + lm] = acc[mt][2][r];
                crow[n0 + 48 + lm] = acc[mt][3][r];
            }
        }
    }
}

// ---------------------------------------------------------------------------
// Setup / pointwise kernels
// ---------------------------------------------------------------------------
// Vectorized f32 -> bf16: 4 floats in (b128 load), 2 packed dwords out
__global__ void cvt_f32_bf16_v4(const float4* __restrict__ in,
                                uint2* __restrict__ out, int n4)
{
    int i = blockIdx.x * 256 + threadIdx.x;
    if (i < n4) {
        float4 v = in[i];
        PackBF lo, hi;
        lo.h[0] = (__bf16)v.x; lo.h[1] = (__bf16)v.y;
        hi.h[0] = (__bf16)v.z; hi.h[1] = (__bf16)v.w;
        out[i] = make_uint2(lo.u, hi.u);
    }
}

__global__ void pack_wcat(const float* __restrict__ Wih,   // [4096, 2048]
                          const float* __restrict__ Whh,   // [4096, 1024]
                          __bf16* __restrict__ out)        // [4096, 3072]
{
    int i = blockIdx.x * 256 + threadIdx.x;  // over G4H*KCAT
    int n = i / KCAT, j = i - n * KCAT;
    float v = (j < H_ + A_) ? Wih[(size_t)n * (H_ + A_) + j]
                            : Whh[(size_t)n * H_ + (j - H_ - A_)];
    out[i] = (__bf16)v;
}

__global__ void bias_sum(const float* __restrict__ bih,
                         const float* __restrict__ bhh,
                         float* __restrict__ bsum)
{
    int i = blockIdx.x * 256 + threadIdx.x;  // G4H
    bsum[i] = bih[i] + bhh[i];
}

// Per step: h (f32) -> h_bf16 (for Wh GEMM) and into xin[:, 2048:3072];
//           x_t (f32) -> xin[:, 0:1024]
__global__ void fill_xh(const float* __restrict__ x_t,
                        const float* __restrict__ h,
                        __bf16* __restrict__ h_bf,
                        __bf16* __restrict__ xin)
{
    int i = blockIdx.x * 256 + threadIdx.x;   // B*H
    int b = i >> 10, j = i & 1023;
    __bf16 hb = (__bf16)h[i];
    h_bf[i] = hb;
    __bf16* row = xin + (size_t)b * KCAT;
    row[j]            = (__bf16)x_t[i];
    row[H_ + A_ + j]  = hb;
}

// scores[b,s] = sum_a v_a[a] * tanh(Wh[b,a] + Uenc[b,s,a])
// Streams ~134MB/step of Uenc: b128 loads, 128 threads x 2 float4 each.
__global__ __launch_bounds__(128)
void scores_kernel(const float* __restrict__ Wh,
                   const float* __restrict__ Uenc,
                   const float* __restrict__ va,
                   float* __restrict__ scores)
{
    __shared__ float red[128];
    int bs = blockIdx.x;          // 0..B*S-1
    int b  = bs >> 8;             // S = 256
    const float4* u4  = (const float4*)(Uenc + (size_t)bs * A_);
    const float4* wh4 = (const float4*)(Wh   + (size_t)b  * A_);
    const float4* va4 = (const float4*)va;
    float s = 0.f;
    #pragma unroll
    for (int i = 0; i < A_ / 4 / 128; ++i) {
        int a = i * 128 + threadIdx.x;
        float4 uu = u4[a], ww = wh4[a], vv = va4[a];
        s += vv.x * fast_tanh(ww.x + uu.x);
        s += vv.y * fast_tanh(ww.y + uu.y);
        s += vv.z * fast_tanh(ww.z + uu.z);
        s += vv.w * fast_tanh(ww.w + uu.w);
    }
    red[threadIdx.x] = s;
    __syncthreads();
    for (int off = 64; off > 0; off >>= 1) {
        if ((int)threadIdx.x < off) red[threadIdx.x] += red[threadIdx.x + off];
        __syncthreads();
    }
    if (threadIdx.x == 0) scores[bs] = red[0];
}

// softmax over S=256 per batch row
__global__ __launch_bounds__(256)
void softmax_kernel(const float* __restrict__ scores, float* __restrict__ attn)
{
    __shared__ float buf[256];
    int b = blockIdx.x, s = threadIdx.x;
    float v = scores[b * S_ + s];
    buf[s] = v;
    __syncthreads();
    for (int off = 128; off > 0; off >>= 1) {
        if (s < off) buf[s] = fmaxf(buf[s], buf[s + off]);
        __syncthreads();
    }
    float mx = buf[0];
    __syncthreads();
    float e = __expf(v - mx);
    buf[s] = e;
    __syncthreads();
    for (int off = 128; off > 0; off >>= 1) {
        if (s < off) buf[s] += buf[s + off];
        __syncthreads();
    }
    attn[b * S_ + s] = e / buf[0];
}

// context[b,a] = sum_s attn[b,s] * att[b,s,a]; write bf16 into xin[:, 1024:2048]
// One block per b: 256 threads, each owns a float4 column slice (A/4 = 256).
__global__ __launch_bounds__(256)
void context_kernel(const float* __restrict__ attn,
                    const float* __restrict__ att,
                    __bf16* __restrict__ xin)
{
    __shared__ float w[256];
    int b = blockIdx.x;
    w[threadIdx.x] = attn[b * S_ + threadIdx.x];
    __syncthreads();
    const float4* base = (const float4*)(att + (size_t)b * S_ * A_) + threadIdx.x;
    float ax = 0.f, ay = 0.f, az = 0.f, aw = 0.f;
    #pragma unroll 4
    for (int s = 0; s < S_; ++s) {
        float4 v = base[(size_t)s * (A_ / 4)];
        float ws = w[s];
        ax += ws * v.x; ay += ws * v.y; az += ws * v.z; aw += ws * v.w;
    }
    PackBF lo, hi;
    lo.h[0] = (__bf16)ax; lo.h[1] = (__bf16)ay;
    hi.h[0] = (__bf16)az; hi.h[1] = (__bf16)aw;
    uint2* dst = (uint2*)(xin + (size_t)b * KCAT + H_);
    dst[threadIdx.x] = make_uint2(lo.u, hi.u);
}

// LSTM pointwise: gate order i, f, g, o (PyTorch LSTMCell)
__global__ __launch_bounds__(256)
void lstm_kernel(const float* __restrict__ gates,
                 float* __restrict__ h, float* __restrict__ c,
                 float* __restrict__ out_t,
                 float* __restrict__ hN, float* __restrict__ cN)
{
    int i = blockIdx.x * 256 + threadIdx.x;   // B*H
    int b = i >> 10, j = i & 1023;
    const float* gr = gates + (size_t)b * G4H;
    float ig = sigmoidf_(gr[j]);
    float fg = sigmoidf_(gr[H_ + j]);
    float gg = fast_tanh(gr[2 * H_ + j]);
    float og = sigmoidf_(gr[3 * H_ + j]);
    float cn = fg * c[i] + ig * gg;
    float hn = og * fast_tanh(cn);
    c[i] = cn;
    h[i] = hn;
    out_t[i] = hn;
    if (hN) { hN[i] = hn; cN[i] = cn; }
}

// ---------------------------------------------------------------------------
extern "C" void kernel_launch(void* const* d_in, const int* in_sizes, int n_in,
                              void* d_out, int out_size, void* d_ws, size_t ws_size,
                              hipStream_t stream)
{
    const float* inputs = (const float*)d_in[0];   // [T,B,H]
    const float* h0     = (const float*)d_in[1];   // [B,H]
    const float* c0     = (const float*)d_in[2];   // [B,H]
    const float* att    = (const float*)d_in[3];   // [B,S,A]
    const float* W_a    = (const float*)d_in[4];   // [A,H]
    const float* U_a_w  = (const float*)d_in[5];   // [A,A]
    const float* U_a_b  = (const float*)d_in[6];   // [A]
    const float* v_a    = (const float*)d_in[7];   // [A]
    const float* W_ih   = (const float*)d_in[8];   // [4H, H+A]
    const float* W_hh   = (const float*)d_in[9];   // [4H, H]
    const float* b_ih   = (const float*)d_in[10];  // [4H]
    const float* b_hh   = (const float*)d_in[11];  // [4H]
    float* out = (float*)d_out;                    // [T,B,H] + [B,H] + [B,H]
    (void)in_sizes; (void)n_in; (void)out_size; (void)ws_size;

    // Workspace carve (256-byte aligned regions)
    char* p = (char*)d_ws;
    auto carve = [&](size_t bytes) -> char* {
        char* r = p;
        p += (bytes + (size_t)255) & ~(size_t)255;
        return r;
    };
    float*  Uenc   = (float*) carve((size_t)B_ * S_ * A_ * 4);
    __bf16* att_bf = (__bf16*)carve((size_t)B_ * S_ * A_ * 2);
    __bf16* Wa_bf  = (__bf16*)carve((size_t)A_ * H_ * 2);
    __bf16* Ua_bf  = (__bf16*)carve((size_t)A_ * A_ * 2);
    __bf16* Wcat   = (__bf16*)carve((size_t)G4H * KCAT * 2);
    float*  bsum   = (float*) carve((size_t)G4H * 4);
    float*  h      = (float*) carve((size_t)B_ * H_ * 4);
    float*  c      = (float*) carve((size_t)B_ * H_ * 4);
    __bf16* h_bf   = (__bf16*)carve((size_t)B_ * H_ * 2);
    float*  Wh     = (float*) carve((size_t)B_ * A_ * 4);
    float*  scores = (float*) carve((size_t)B_ * S_ * 4);
    float*  attn   = (float*) carve((size_t)B_ * S_ * 4);
    __bf16* xin    = (__bf16*)carve((size_t)B_ * KCAT * 2);
    float*  gates  = (float*) carve((size_t)B_ * G4H * 4);

    // State init (capture-safe d2d async copies)
    hipMemcpyAsync(h, h0, (size_t)B_ * H_ * 4, hipMemcpyDeviceToDevice, stream);
    hipMemcpyAsync(c, c0, (size_t)B_ * H_ * 4, hipMemcpyDeviceToDevice, stream);

    // One-time conversions / packing (vectorized b128 reads)
    const int nAtt4 = (B_ * S_ * A_) / 4;
    cvt_f32_bf16_v4<<<nAtt4 / 256, 256, 0, stream>>>(
        (const float4*)att, (uint2*)att_bf, nAtt4);
    cvt_f32_bf16_v4<<<(A_ * H_ / 4) / 256, 256, 0, stream>>>(
        (const float4*)W_a, (uint2*)Wa_bf, A_ * H_ / 4);
    cvt_f32_bf16_v4<<<(A_ * A_ / 4) / 256, 256, 0, stream>>>(
        (const float4*)U_a_w, (uint2*)Ua_bf, A_ * A_ / 4);
    pack_wcat<<<(G4H * KCAT) / 256, 256, 0, stream>>>(W_ih, W_hh, Wcat);
    bias_sum<<<G4H / 256, 256, 0, stream>>>(b_ih, b_hh, bsum);

    // Uenc[b,s,:] = att[b,s,:] @ U_a_w^T + U_a_b   (M=32768, N=1024, K=1024)
    // Tall GEMM -> MT=2 (32x64 wave tile, 2x B reuse), 512x4 = 2048 blocks.
    gemm_bf16_nt<2><<<dim3(A_ / 256, (B_ * S_) / 64), 256, 0, stream>>>(
        (const uint32_t*)att_bf, (const uint32_t*)Ua_bf, U_a_b, Uenc,
        B_ * S_, A_, A_);

    for (int t = 0; t < T_; ++t) {
        fill_xh<<<(B_ * H_) / 256, 256, 0, stream>>>(
            inputs + (size_t)t * B_ * H_, h, h_bf, xin);

        // Wh = h @ W_a^T   (M=128, N=1024, K=1024) -> MT=1, 16 blocks
        gemm_bf16_nt<1><<<dim3(A_ / 256, B_ / 32), 256, 0, stream>>>(
            (const uint32_t*)h_bf, (const uint32_t*)Wa_bf, nullptr, Wh,
            B_, A_, H_);

        scores_kernel<<<B_ * S_, 128, 0, stream>>>(Wh, Uenc, v_a, scores);
        softmax_kernel<<<B_, 256, 0, stream>>>(scores, attn);
        context_kernel<<<B_, 256, 0, stream>>>(attn, att, xin);

        // gates = [x|context|h] @ [W_ih|W_hh]^T + (b_ih+b_hh)
        // (M=128, N=4096, K=3072) -> MT=1, 64 blocks
        gemm_bf16_nt<1><<<dim3(G4H / 256, B_ / 32), 256, 0, stream>>>(
            (const uint32_t*)xin, (const uint32_t*)Wcat, bsum, gates,
            B_, G4H, KCAT);

        const bool last = (t == T_ - 1);
        lstm_kernel<<<(B_ * H_) / 256, 256, 0, stream>>>(
            gates, h, c, out + (size_t)t * B_ * H_,
            last ? out + (size_t)T_ * B_ * H_ : nullptr,
            last ? out + (size_t)T_ * B_ * H_ + (size_t)B_ * H_ : nullptr);
    }
}
